// ALSHConv2d_81896436400439
// MI455X (gfx1250) — matrix operand order repaired and checked by reading, not compiled, not run on hardware
//
#include <hip/hip_runtime.h>

// ---------------------------------------------------------------------------
// Implicit-GEMM 3x3 conv (NHWC, 64->64) for MI455X / gfx1250, v2.
//   Math path: V_WMMA_F32_16X16X32_BF16 (bf16 A/B, f32 accumulate).
//   v2 change: each wave now owns 64 pixels (4 B-subtiles), so every A
//   (weight) fragment feeds 4 WMMAs. LDS read traffic drops from
//   1.25 KB/WMMA (~1.13 GB total, LDS-bound ~9us) to 0.5 KB/WMMA
//   (~450 MB, ~3.7us) -- below the 4.4us HBM roofline (103 MB @ 23.3 TB/s).
// ---------------------------------------------------------------------------

typedef __attribute__((ext_vector_type(16))) __bf16 v16bf;
typedef __attribute__((ext_vector_type(8)))  float  v8f;

#define C_IN    64
#define C_OUT   64
#define HH      112
#define WW      112
#define NBATCH  16
#define KTOT    576          // 3*3*64
#define KSTEPS  18           // 576 / 32
#define MT      4            // C_OUT / 16 output-channel tiles
#define ST      4            // pixel subtiles per wave (4 x 16 = 64 pixels)
#define WAVES   7
#define TPB     (WAVES * 32) // 224 threads; 7 waves * 64 px = 448 px = 4 rows
#define ROWS    4            // output rows per workgroup
#define SROWS   (ROWS + 2)   // staged input rows (with halo)
#define PADW    114          // 112 + 2 halo columns

// LDS layout (bytes)
#define LDS_W_BYTES (MT * KSTEPS * 32 * 16 * 2)   // 73728: packed bf16 weights
#define LDS_X_OFF   LDS_W_BYTES
#define LDS_X_BYTES (SROWS * PADW * C_IN * 2)     // 87552: 6 padded input rows
#define LDS_TOTAL   (LDS_W_BYTES + LDS_X_BYTES)   // 161280 (2 WGs / WGP)

struct __align__(8) bf16x4 { __bf16 a, b, c, d; };

// ---------------------------------------------------------------------------
// Prep: fp32 weights (O, kh, kw, ci) -> bf16, prepacked in WMMA A-fragment
// order.  A (16x32 bf16) layout this kernel uses (contiguous K, as B): lanes 0-15 hold row M=lane,
// K = {0..15}; lanes 16-31 hold row M=lane-16, K = {16..31}.
// Flat OHWI weight index for row m is m*576 + kk.
// ---------------------------------------------------------------------------
__global__ void pack_weights_bf16(const float* __restrict__ w,
                                  __bf16* __restrict__ wp) {
    int idx  = blockIdx.x * blockDim.x + threadIdx.x;   // 0 .. 36863
    int e    = idx & 15;
    int lane = (idx >> 4) & 31;
    int ks   = (idx >> 9) % KSTEPS;
    int mt   = (idx >> 9) / KSTEPS;
    int m    = mt * 16 + (lane & 15);
    int kk   = ks * 32 + ((lane < 16) ? 0 : 16) + e;
    wp[idx]  = (__bf16)w[m * KTOT + kk];
}

// ---------------------------------------------------------------------------
// Main: one workgroup per (n, 4-row strip). 7 waves x 64 pixels.
// ---------------------------------------------------------------------------
__global__ void __launch_bounds__(TPB)
conv3x3_wmma_bf16(const float* __restrict__ x,
                  const __bf16* __restrict__ wp,
                  float* __restrict__ out) {
    extern __shared__ char smem[];
    __bf16* lw = (__bf16*)smem;                 // packed weights
    __bf16* lx = (__bf16*)(smem + LDS_X_OFF);   // padded bf16 input rows

    const int tid = threadIdx.x;
    const int blk = blockIdx.x;
    const int n   = blk / 28;
    const int h0  = (blk % 28) * ROWS;          // first output row of strip

    // ---- stage packed weights into LDS (raw 16B copies) -------------------
    {
        const uint4* src = (const uint4*)wp;
        uint4*       dst = (uint4*)lw;
        #pragma unroll 2
        for (int i = tid; i < LDS_W_BYTES / 16; i += TPB) dst[i] = src[i];
    }

    // ---- zero-fill padded input staging (covers halo + OOB rows) ----------
    {
        uint4 z; z.x = z.y = z.z = z.w = 0u;
        uint4* xz = (uint4*)lx;
        #pragma unroll 2
        for (int i = tid; i < LDS_X_BYTES / 16; i += TPB) xz[i] = z;
    }
    __syncthreads();

    // ---- load valid input rows h0-1 .. h0+4, convert fp32 -> bf16 ---------
    // Each row: 112 cols * 64 ch = 1792 float4 chunks; 6*1792 total.
    for (int v = tid; v < SROWS * 1792; v += TPB) {
        int r  = v / 1792;
        int gh = h0 - 1 + r;
        if (gh < 0 || gh >= HH) continue;        // stays zero (padding row)
        int rem = v % 1792;
        int cw  = rem >> 4;                      // column 0..111
        int c4  = (rem & 15) * 4;                // channel 0,4,..,60
        const float4 f =
            *(const float4*)(x + (((long)n * HH + gh) * WW + cw) * C_IN + c4);
        bf16x4 s;
        s.a = (__bf16)f.x; s.b = (__bf16)f.y;
        s.c = (__bf16)f.z; s.d = (__bf16)f.w;
        *(bf16x4*)(lx + (r * PADW + (cw + 1)) * C_IN + c4) = s;
    }
    __syncthreads();

    // ---- WMMA compute: 64 pixels x 64 channels per wave -------------------
    const int lane = tid & 31;
    const int wv   = tid >> 5;            // 0..6
    const int half = lane >> 4;           // 0 / 1 (lane group)
    const int nn   = lane & 15;

    // Subtile sub: 16 pixels at strip-row rr, columns cb..cb+15.
    // base = 16*(wv*4+sub); 112 = 7*16 so each subtile sits in one row.
    int rr[ST], cb[ST];
    #pragma unroll
    for (int s = 0; s < ST; ++s) {
        int bidx = wv * ST + s;           // 0..27
        rr[s] = bidx / 7;                 // strip row 0..3
        cb[s] = (bidx % 7) * 16;          // column base
    }

    v8f acc[MT][ST] = {};                 // 16 f32 accumulator tiles

    for (int ks = 0; ks < KSTEPS; ++ks) {
        // This lane-group's 16 consecutive K values -> (kh, kw, c0).
        // 16-aligned start => run never crosses a (kh,kw) boundary.
        int kb = ks * 32 + half * 16;
        int kh = kb / 192;
        int kr = kb - kh * 192;
        int kw = kr >> 6;
        int c0 = kr & 63;

        v16bf bfr[ST];
        #pragma unroll
        for (int s = 0; s < ST; ++s) {
            bfr[s] = *(const v16bf*)(lx +
                ((rr[s] + kh) * PADW + (cb[s] + nn + kw)) * C_IN + c0);
        }

        #pragma unroll
        for (int m = 0; m < MT; ++m) {
            const v16bf afrag =
                *(const v16bf*)(lw + ((m * KSTEPS + ks) * 32 + lane) * 16);
            #pragma unroll
            for (int s = 0; s < ST; ++s) {
                acc[m][s] = __builtin_amdgcn_wmma_f32_16x16x32_bf16(
                    false, afrag, false, bfr[s], (short)0, acc[m][s],
                    false, false);
            }
        }
    }

    // ---- store: lane holds 8 contiguous channels per (m, subtile) ---------
    #pragma unroll
    for (int s = 0; s < ST; ++s) {
        float* op = out +
            (((long)n * HH + (h0 + rr[s])) * WW + (cb[s] + nn)) * C_OUT;
        #pragma unroll
        for (int m = 0; m < MT; ++m) {
            int c = m * 16 + half * 8;    // channel base for this lane
            float4 lo, hi;
            lo.x = acc[m][s][0]; lo.y = acc[m][s][1];
            lo.z = acc[m][s][2]; lo.w = acc[m][s][3];
            hi.x = acc[m][s][4]; hi.y = acc[m][s][5];
            hi.z = acc[m][s][6]; hi.w = acc[m][s][7];
            *(float4*)(op + c)     = lo;
            *(float4*)(op + c + 4) = hi;
        }
    }
}

// ---------------------------------------------------------------------------
extern "C" void kernel_launch(void* const* d_in, const int* in_sizes, int n_in,
                              void* d_out, int out_size, void* d_ws,
                              size_t ws_size, hipStream_t stream) {
    const float* x    = (const float*)d_in[0];   // (16,112,112,64) f32
    const float* kern = (const float*)d_in[1];   // (64,3,3,64)     f32
    float*       out  = (float*)d_out;           // (16,112,112,64) f32
    __bf16*      wp   = (__bf16*)d_ws;           // 73728 B packed weights

    // 1) pack weights: 4*18*32*16 = 36864 elements
    pack_weights_bf16<<<144, 256, 0, stream>>>(kern, wp);

    // 2) conv: one block per (image, 4-row strip)
    conv3x3_wmma_bf16<<<NBATCH * 28, TPB, LDS_TOTAL, stream>>>(x, wp, out);
}
